// CrossEfficientAttention_62079457297008
// MI455X (gfx1250) — matrix operand order repaired and checked by reading, not compiled, not run on hardware
//
#include <hip/hip_runtime.h>
#include <hip/hip_bf16.h>

// ---------------------------------------------------------------------------
// CrossEfficientAttention for MI455X (gfx1250), wave32 + WMMA bf16.
//   n=8, C=256, H=W=80 -> L=6400, HEADS=8 (32 ch/head).
// Fusion: out = A[n] @ softmax_ch(Wq@X+bq) + br + X,
//         A[n] = Wr @ blockdiag(context[n,h]^T),
//         context[n,h] = softmax_L(Wk@Xk+bk) @ (Wv@Xv+bv)^T  (per head).
// Memory-bound (~28.5 GF vs >=210MB fp32 @ 23.3 TB/s): all GEMMs on
// v_wmma_f32_16x16x32_bf16; Q never touches HBM.
//
// LDS tiles are stored FRAGMENT-PACKED: [tile][lane][16 bf16] in exactly the
// WMMA per-lane register order, so every operand load is 2x ds_load_b128 and
// every staging write is a 16B/4B vector store (no scalar u16 LDS traffic).
// ---------------------------------------------------------------------------

typedef __bf16 bf16_t;
typedef __attribute__((ext_vector_type(16))) __bf16 v16bf;
typedef __attribute__((ext_vector_type(8)))  __bf16 v8bf;
typedef __attribute__((ext_vector_type(2)))  __bf16 v2bf;
typedef __attribute__((ext_vector_type(8)))  float  v8f;

#define CCH  256
#define LSP  6400
#define NB   8
#define KT   32

// Packed-tile fragment load: lane's 16 bf16 are contiguous -> 2x ds_load_b128.
__device__ __forceinline__ v16bf load_frag(const bf16_t* s, int tile, int lane) {
  const v8bf* p = (const v8bf*)(s + ((tile * 32 + lane) << 4));
  v8bf lo = p[0];
  v8bf hi = p[1];
  return __builtin_shufflevector(lo, hi, 0, 1, 2, 3, 4, 5, 6, 7,
                                 8, 9, 10, 11, 12, 13, 14, 15);
}

__device__ __forceinline__ v8bf cvt8(float4 a, float4 b) {
  v8bf r;
  r[0] = (bf16_t)a.x; r[1] = (bf16_t)a.y; r[2] = (bf16_t)a.z; r[3] = (bf16_t)a.w;
  r[4] = (bf16_t)b.x; r[5] = (bf16_t)b.y; r[6] = (bf16_t)b.z; r[7] = (bf16_t)b.w;
  return r;
}
__device__ __forceinline__ v8bf cvt8exp(float4 a, float4 b, float rm) {
  v8bf r;
  r[0] = (bf16_t)__expf(a.x - rm); r[1] = (bf16_t)__expf(a.y - rm);
  r[2] = (bf16_t)__expf(a.z - rm); r[3] = (bf16_t)__expf(a.w - rm);
  r[4] = (bf16_t)__expf(b.x - rm); r[5] = (bf16_t)__expf(b.y - rm);
  r[6] = (bf16_t)__expf(b.z - rm); r[7] = (bf16_t)__expf(b.w - rm);
  return r;
}

// Packed destination for an A-matrix K-octet (8 consecutive K at octet j of a
// 16-row tile): lane' = row + (j&2)*8, elem base = (j&1)?8:0  -> 16B store.
__device__ __forceinline__ bf16_t* a_oct_dst(bf16_t* s, int tile, int row, int j) {
  return s + (((tile * 32) + row + ((j & 2) ? 16 : 0)) << 4) + ((j & 1) ? 8 : 0);
}

#define WMMA_BF16(a, b, c) \
  __builtin_amdgcn_wmma_f32_16x16x32_bf16(false, (a), false, (b), (short)0, (c), false, false)

// ---------------------------------------------------------------------------
// Kernel 1: Out[n,o,l] = sum_c W[o,c]*X[n,c,l] + bias[o]   (used for K and V)
// Tile 128(o) x 64(l), 8 waves as 4x2 of 32x32, K-loop 256 step 32.
// ---------------------------------------------------------------------------
__global__ __launch_bounds__(256) void proj_gemm_kernel(
    const float* __restrict__ X, const float* __restrict__ W,
    const float* __restrict__ bias, float* __restrict__ Out) {
  __shared__ bf16_t sA[8 * 512];  // 8 row-tiles of packed [32][16]  (8KB)
  __shared__ bf16_t sB[4 * 512];  // 4 col-tiles packed              (4KB)
  const int lt = blockIdx.x, mt = blockIdx.y, n = blockIdx.z;
  const int tid = threadIdx.x, lane = tid & 31, wave = tid >> 5;
  const int l0 = lt * 64, m0 = mt * 128;
  const int wm = wave >> 1, wn = wave & 1;
  v8f acc[2][2] = {};
  for (int k0 = 0; k0 < CCH; k0 += KT) {
    // ---- stage W tile (128 rows x 32 k) as packed octets: 2 per thread ----
#pragma unroll
    for (int jj = 0; jj < 2; ++jj) {
      const int o = tid + jj * 256;       // 512 octets
      const int row = o >> 2, j = o & 3;  // row 0..127, K-octet 0..3
      const float* wp = W + (size_t)(m0 + row) * CCH + k0 + j * 8;
      float4 f0 = *(const float4*)wp;
      float4 f1 = *(const float4*)(wp + 4);
      *(v8bf*)a_oct_dst(sA, row >> 4, row & 15, j) = cvt8(f0, f1);
    }
    // ---- stage X tile (32 k x 64 l): k-pair per thread, v2bf stores ----
    {
      const int p = tid >> 4, q = tid & 15;  // k-pair, col-quad
      const float* xr = X + ((size_t)n * CCH + k0 + 2 * p) * LSP + l0 + q * 4;
      float4 x0 = *(const float4*)xr;
      float4 x1 = *(const float4*)(xr + LSP);
      const float c0[4] = {x0.x, x0.y, x0.z, x0.w};
      const float c1[4] = {x1.x, x1.y, x1.z, x1.w};
      const int elem = (2 * p) & 15, lh = (2 * p) & 16;
#pragma unroll
      for (int jq = 0; jq < 4; ++jq) {
        const int col = q * 4 + jq;
        v2bf pr; pr[0] = (bf16_t)c0[jq]; pr[1] = (bf16_t)c1[jq];
        *(v2bf*)(sB + (((col >> 4) * 32 + (col & 15) + lh) << 4) + elem) = pr;
      }
    }
    if (k0 + KT < CCH)
      __builtin_prefetch(&X[((size_t)n * CCH + (k0 + KT + (tid >> 3))) * LSP + l0 + ((tid & 7) << 3)], 0, 1);
    __syncthreads();
    v16bf a0 = load_frag(sA, wm * 2, lane);
    v16bf a1 = load_frag(sA, wm * 2 + 1, lane);
#pragma unroll
    for (int ni = 0; ni < 2; ++ni) {
      v16bf b = load_frag(sB, wn * 2 + ni, lane);
      acc[0][ni] = WMMA_BF16(a0, b, acc[0][ni]);
      acc[1][ni] = WMMA_BF16(a1, b, acc[1][ni]);
    }
    __syncthreads();
  }
  const int colw = lane & 15, rowoff = (lane & 16) ? 8 : 0;
#pragma unroll
  for (int mi = 0; mi < 2; ++mi)
#pragma unroll
    for (int ni = 0; ni < 2; ++ni) {
      const int col = l0 + wn * 32 + ni * 16 + colw;
#pragma unroll
      for (int r = 0; r < 8; ++r) {
        const int row = m0 + wm * 32 + mi * 16 + rowoff + r;
        Out[((size_t)n * CCH + row) * LSP + col] = acc[mi][ni][r] + bias[row];
      }
    }
}

// ---------------------------------------------------------------------------
// Kernel 2: per-row max and sum(exp(x-max)) over L for the k-softmax.
// ---------------------------------------------------------------------------
__global__ __launch_bounds__(256) void kstats_kernel(const float* __restrict__ Kp,
                                                     float* __restrict__ rowmax,
                                                     float* __restrict__ rowsum) {
  __shared__ float red[256];
  const int row = blockIdx.x, tid = threadIdx.x;
  const float* p = Kp + (size_t)row * LSP;
  float vals[25];
  float m = -3.0e38f;
#pragma unroll
  for (int i = 0; i < 25; ++i) { vals[i] = p[tid + i * 256]; m = fmaxf(m, vals[i]); }
  red[tid] = m; __syncthreads();
  for (int s = 128; s > 0; s >>= 1) {
    if (tid < s) red[tid] = fmaxf(red[tid], red[tid + s]);
    __syncthreads();
  }
  m = red[0]; __syncthreads();
  float s = 0.f;
#pragma unroll
  for (int i = 0; i < 25; ++i) s += __expf(vals[i] - m);
  red[tid] = s; __syncthreads();
  for (int st = 128; st > 0; st >>= 1) {
    if (tid < st) red[tid] += red[tid + st];
    __syncthreads();
  }
  if (tid == 0) { rowmax[row] = m; rowsum[row] = red[0]; }
}

// ---------------------------------------------------------------------------
// Kernel 3: context[n,h] = (exp(k-max) @ v^T) / rowsum   (32x32, K=6400)
// One block per (n,h); 8 waves own 800-col L-slices; per-wave packed LDS
// staging; deterministic sequential cross-wave reduction.
// ---------------------------------------------------------------------------
__global__ __launch_bounds__(256) void context_kernel(
    const float* __restrict__ Kp, const float* __restrict__ Vp,
    const float* __restrict__ rowmax, const float* __restrict__ rowsum,
    float* __restrict__ ctx_out) {
  __shared__ bf16_t kbuf[8][2 * 512];  // per-wave packed A (2 row-tiles)
  __shared__ bf16_t vbuf[8][2 * 512];  // per-wave packed B (2 col-tiles)
  __shared__ float  ctxred[32 * 32];
  const int n = blockIdx.x >> 3, h = blockIdx.x & 7;
  const int tid = threadIdx.x, lane = tid & 31, wave = tid >> 5;
  const size_t base = ((size_t)n * CCH + h * 32) * LSP;
  const float rm = rowmax[n * CCH + h * 32 + lane];
  bf16_t* kb = kbuf[wave];
  bf16_t* vb = vbuf[wave];
  const int tile = lane >> 4, rl = lane & 15;
  v8f acc[2][2] = {};
  for (int c = 0; c < 25; ++c) {
    const int l0 = (wave * 25 + c) * 32;
    const float* kr = Kp + base + (size_t)lane * LSP + l0;
    const float* vr = Vp + base + (size_t)lane * LSP + l0;
    // A (=exp(k-max)) row `lane`: 4 packed K-octets, 16B stores
#pragma unroll
    for (int j = 0; j < 4; ++j) {
      float4 f0 = *(const float4*)(kr + j * 8);
      float4 f1 = *(const float4*)(kr + j * 8 + 4);
      *(v8bf*)a_oct_dst(kb, tile, rl, j) = cvt8exp(f0, f1, rm);
    }
    // B (=v^T) column `lane`: two 16-K halves -> contiguous 32B each
#pragma unroll
    for (int half = 0; half < 2; ++half) {
      float4 f0 = *(const float4*)(vr + half * 16);
      float4 f1 = *(const float4*)(vr + half * 16 + 4);
      float4 f2 = *(const float4*)(vr + half * 16 + 8);
      float4 f3 = *(const float4*)(vr + half * 16 + 12);
      bf16_t* dst = vb + ((tile * 32 + rl + half * 16) << 4);
      *(v8bf*)dst = cvt8(f0, f1);
      *(v8bf*)(dst + 8) = cvt8(f2, f3);
    }
    __builtin_amdgcn_wave_barrier();  // wave-synchronous LDS staging
    v16bf a0 = load_frag(kb, 0, lane);
    v16bf a1 = load_frag(kb, 1, lane);
#pragma unroll
    for (int ni = 0; ni < 2; ++ni) {
      v16bf b = load_frag(vb, ni, lane);
      acc[0][ni] = WMMA_BF16(a0, b, acc[0][ni]);
      acc[1][ni] = WMMA_BF16(a1, b, acc[1][ni]);
    }
    __builtin_amdgcn_wave_barrier();
  }
  for (int i = tid; i < 1024; i += 256) ctxred[i] = 0.f;
  __syncthreads();
  const int colw = lane & 15, rowoff = (lane & 16) ? 8 : 0;
  for (int w = 0; w < 8; ++w) {  // fixed order -> deterministic
    if (wave == w) {
#pragma unroll
      for (int mi = 0; mi < 2; ++mi)
#pragma unroll
        for (int ni = 0; ni < 2; ++ni)
#pragma unroll
          for (int r = 0; r < 8; ++r)
            ctxred[(mi * 16 + rowoff + r) * 32 + ni * 16 + colw] += acc[mi][ni][r];
    }
    __syncthreads();
  }
  for (int i = tid; i < 1024; i += 256) {
    const int ck = i >> 5, cv = i & 31;
    ctx_out[(((size_t)n * 8 + h) * 32 + ck) * 32 + cv] =
        ctxred[i] / rowsum[n * CCH + h * 32 + ck];
  }
}

// ---------------------------------------------------------------------------
// Kernel 4: A[n][co][h*32+ck] = sum_cv Wr[co][h*32+cv] * ctx[n][h][ck][cv]
// ---------------------------------------------------------------------------
__global__ __launch_bounds__(256) void amat_kernel(const float* __restrict__ Wr,
                                                   const float* __restrict__ ctx,
                                                   bf16_t* __restrict__ A) {
  const int n = blockIdx.x, co = threadIdx.x;
  for (int h = 0; h < 8; ++h) {
    const float* c = ctx + (((size_t)n * 8 + h) * 32) * 32;
    const float* w = Wr + (size_t)co * CCH + h * 32;
    for (int ck = 0; ck < 32; ++ck) {
      float s = 0.f;
#pragma unroll
      for (int cv = 0; cv < 32; ++cv) s += w[cv] * c[ck * 32 + cv];
      A[((size_t)n * CCH + co) * CCH + h * 32 + ck] = (bf16_t)s;
    }
  }
}

// ---------------------------------------------------------------------------
// Kernel 5: fused  Q=Wq@X+bq -> head-channel softmax -> A[n]@Qsoft
//           -> +br +residual.  Block per (l-tile 64, n); wave w owns head w.
// LDS 52KB: sA 16KB (Wq packed; reused as per-wave A[n] buffers in phase 2),
// sB 4KB (X tile), sQ 32KB (Qsoft packed as phase-2 B operand).
// ---------------------------------------------------------------------------
__global__ __launch_bounds__(256) void fused_q_attn_out_kernel(
    const float* __restrict__ Xq, const float* __restrict__ Wq,
    const float* __restrict__ bq, const bf16_t* __restrict__ Amat,
    const float* __restrict__ br, float* __restrict__ Out) {
  __shared__ bf16_t sA[16 * 512];  // 16 row-tiles (Wq) / 8 waves x 2 tiles (A[n])
  __shared__ bf16_t sB[4 * 512];   // X tile
  __shared__ bf16_t sQ[32 * 512];  // Qsoft: [kblock 8][coltile 4] packed tiles
  const int lt = blockIdx.x, n = blockIdx.y;
  const int tid = threadIdx.x, lane = tid & 31, wave = tid >> 5;
  const int l0 = lt * 64;
  const int colw = lane & 15, rowoff = (lane & 16) ? 8 : 0;
  v8f acc[2][4] = {};

  // ---- Phase 1: Q tile = Wq @ X + bq ----
  for (int k0 = 0; k0 < CCH; k0 += KT) {
#pragma unroll
    for (int jj = 0; jj < 4; ++jj) {  // 1024 Wq octets
      const int o = tid + jj * 256;
      const int row = o >> 2, j = o & 3;
      const float* wp = Wq + (size_t)row * CCH + k0 + j * 8;
      float4 f0 = *(const float4*)wp;
      float4 f1 = *(const float4*)(wp + 4);
      *(v8bf*)a_oct_dst(sA, row >> 4, row & 15, j) = cvt8(f0, f1);
    }
    {
      const int p = tid >> 4, q = tid & 15;
      const float* xr = Xq + ((size_t)n * CCH + k0 + 2 * p) * LSP + l0 + q * 4;
      float4 x0 = *(const float4*)xr;
      float4 x1 = *(const float4*)(xr + LSP);
      const float c0[4] = {x0.x, x0.y, x0.z, x0.w};
      const float c1[4] = {x1.x, x1.y, x1.z, x1.w};
      const int elem = (2 * p) & 15, lh = (2 * p) & 16;
#pragma unroll
      for (int jq = 0; jq < 4; ++jq) {
        const int col = q * 4 + jq;
        v2bf pr; pr[0] = (bf16_t)c0[jq]; pr[1] = (bf16_t)c1[jq];
        *(v2bf*)(sB + (((col >> 4) * 32 + (col & 15) + lh) << 4) + elem) = pr;
      }
    }
    __syncthreads();
    v16bf a0 = load_frag(sA, wave * 2, lane);
    v16bf a1 = load_frag(sA, wave * 2 + 1, lane);
#pragma unroll
    for (int ni = 0; ni < 4; ++ni) {
      v16bf b = load_frag(sB, ni, lane);
      acc[0][ni] = WMMA_BF16(a0, b, acc[0][ni]);
      acc[1][ni] = WMMA_BF16(a1, b, acc[1][ni]);
    }
    __syncthreads();
  }
  // bias
#pragma unroll
  for (int mi = 0; mi < 2; ++mi)
#pragma unroll
    for (int r = 0; r < 8; ++r) {
      const float bv = bq[wave * 32 + mi * 16 + rowoff + r];
#pragma unroll
      for (int ni = 0; ni < 4; ++ni) acc[mi][ni][r] += bv;
    }
  // softmax over the head's 32 channels: lane halves (lane, lane^16) hold
  // complementary rows of each column -> one shfl_xor(16) per reduction.
#pragma unroll
  for (int ni = 0; ni < 4; ++ni) {
    float m = -3.0e38f;
#pragma unroll
    for (int mi = 0; mi < 2; ++mi)
#pragma unroll
      for (int r = 0; r < 8; ++r) m = fmaxf(m, acc[mi][ni][r]);
    m = fmaxf(m, __shfl_xor(m, 16, 32));
    float s = 0.f;
#pragma unroll
    for (int mi = 0; mi < 2; ++mi)
#pragma unroll
      for (int r = 0; r < 8; ++r) {
        const float e = __expf(acc[mi][ni][r] - m);
        acc[mi][ni][r] = e;
        s += e;
      }
    s += __shfl_xor(s, 16, 32);
    const float inv = 1.0f / s;
#pragma unroll
    for (int mi = 0; mi < 2; ++mi)
#pragma unroll
      for (int r = 0; r < 8; ++r) acc[mi][ni][r] *= inv;
  }
  // stash Qsoft into packed phase-2 B tiles: each (mi,ni) is one 16B store
#pragma unroll
  for (int mi = 0; mi < 2; ++mi)
#pragma unroll
    for (int ni = 0; ni < 4; ++ni) {
      v8bf qv;
#pragma unroll
      for (int r = 0; r < 8; ++r) qv[r] = (bf16_t)acc[mi][ni][r];
      *(v8bf*)(sQ + (((wave * 4 + ni) * 32 + colw + 16 * mi) << 4) + rowoff) = qv;
    }
  __syncthreads();

  // ---- Phase 2: Out tile = A[n] @ Qsoft ----
#pragma unroll
  for (int mi = 0; mi < 2; ++mi)
#pragma unroll
    for (int ni = 0; ni < 4; ++ni) acc[mi][ni] = (v8f){};
  bf16_t* aW = sA + wave * 1024;  // per-wave 2 packed row-tiles
  const int trow = lane >> 4, rlw = lane & 15;
  for (int k0 = 0; k0 < CCH; k0 += KT) {
    const v8bf* g = (const v8bf*)(Amat + ((size_t)n * CCH + wave * 32 + lane) * CCH + k0);
#pragma unroll
    for (int j = 0; j < 4; ++j)
      *(v8bf*)a_oct_dst(aW, trow, rlw, j) = g[j];
    __builtin_amdgcn_wave_barrier();
    v16bf a0 = load_frag(aW, 0, lane);
    v16bf a1 = load_frag(aW, 1, lane);
    const int kb2 = k0 >> 5;
#pragma unroll
    for (int ni = 0; ni < 4; ++ni) {
      v16bf b = load_frag(sQ, kb2 * 4 + ni, lane);
      acc[0][ni] = WMMA_BF16(a0, b, acc[0][ni]);
      acc[1][ni] = WMMA_BF16(a1, b, acc[1][ni]);
    }
    __builtin_amdgcn_wave_barrier();
  }
  // epilogue: + br + residual
#pragma unroll
  for (int mi = 0; mi < 2; ++mi)
#pragma unroll
    for (int ni = 0; ni < 4; ++ni) {
      const int col = l0 + ni * 16 + colw;
#pragma unroll
      for (int r = 0; r < 8; ++r) {
        const int row = wave * 32 + mi * 16 + rowoff + r;
        const size_t idx = ((size_t)n * CCH + row) * LSP + col;
        Out[idx] = acc[mi][ni][r] + br[row] + Xq[idx];
      }
    }
}

// ---------------------------------------------------------------------------
extern "C" void kernel_launch(void* const* d_in, const int* in_sizes, int n_in,
                              void* d_out, int out_size, void* d_ws, size_t ws_size,
                              hipStream_t stream) {
  const float* qf = (const float*)d_in[0];
  const float* kf = (const float*)d_in[1];
  const float* vf = (const float*)d_in[2];
  const float* Wq = (const float*)d_in[3];
  const float* bq = (const float*)d_in[4];
  const float* Wk = (const float*)d_in[5];
  const float* bk = (const float*)d_in[6];
  const float* Wv = (const float*)d_in[7];
  const float* bv = (const float*)d_in[8];
  const float* Wr = (const float*)d_in[9];
  const float* br = (const float*)d_in[10];
  float* out = (float*)d_out;

  const size_t proj_elems = (size_t)NB * CCH * LSP;  // 13,107,200
  float*  ws_k   = (float*)d_ws;
  float*  ws_v   = ws_k + proj_elems;
  float*  rowmax = ws_v + proj_elems;                // NB*CCH = 2048
  float*  rowsum = rowmax + NB * CCH;
  float*  ctx    = rowsum + NB * CCH;                // 8*8*32*32 = 65536
  bf16_t* Abf    = (bf16_t*)(ctx + NB * 8 * 32 * 32);  // 8*256*256 bf16

  dim3 gproj(LSP / 64, CCH / 128, NB);  // (100, 2, 8)
  proj_gemm_kernel<<<gproj, 256, 0, stream>>>(kf, Wk, bk, ws_k);
  proj_gemm_kernel<<<gproj, 256, 0, stream>>>(vf, Wv, bv, ws_v);
  kstats_kernel<<<NB * CCH, 256, 0, stream>>>(ws_k, rowmax, rowsum);
  context_kernel<<<NB * 8, 256, 0, stream>>>(ws_k, ws_v, rowmax, rowsum, ctx);
  amat_kernel<<<NB, 256, 0, stream>>>(Wr, ctx, Abf);
  fused_q_attn_out_kernel<<<dim3(LSP / 64, NB), 256, 0, stream>>>(
      qf, Wq, bq, Abf, br, out);
}
